// DynamicFilterConv_47966194762224
// MI455X (gfx1250) — matrix operand order repaired and checked by reading, not compiled, not run on hardware
//
#include <hip/hip_runtime.h>

typedef _Float16 h16v __attribute__((ext_vector_type(16)));
typedef _Float16 h8   __attribute__((ext_vector_type(8)));
typedef float    f8v  __attribute__((ext_vector_type(8)));

#define B_    16
#define CIN_  64
#define COUT_ 64
#define NCOND 8
#define H_    128
#define W_    128
#define KP    72              // padded cin stride (halves) -> 144B: 16B-aligned, 36-bank stride
#define WPB   (COUT_*9*KP)    // halves of effective filter per batch = 41472
#define TPB   512             // 16 wave32 per workgroup

// ---------------------------------------------------------------------------
// Kernel A: softmax gate + effective per-batch filters/bias
//   wEff layout: [b][cout(64)][tap(9)][KP(72)]  (f16, cin padded with zeros)
//   bEff layout: [b][cout]                      (f32)
// ---------------------------------------------------------------------------
__global__ __launch_bounds__(256) void dfc_weights(
    const float* __restrict__ cond,   // [B, CIN]
    const float* __restrict__ filt_w, // [N, COUT, CIN, 3, 3]
    const float* __restrict__ filt_b, // [N, COUT]
    const float* __restrict__ sel_w,  // [N, CIN]
    const float* __restrict__ sel_b,  // [N]
    _Float16* __restrict__ wEff,
    float* __restrict__ bEff) {
  const int b = blockIdx.x;
  const int tid = threadIdx.x;
  __shared__ float slog[NCOND];
  __shared__ float sw[NCOND];

  if (tid < NCOND) {
    float acc = sel_b[tid];
    const float* c = cond + b * CIN_;
    const float* w = sel_w + tid * CIN_;
    for (int i = 0; i < CIN_; ++i) acc += c[i] * w[i];
    slog[tid] = acc;
  }
  __syncthreads();
  if (tid == 0) {
    float m = slog[0];
    for (int n = 1; n < NCOND; ++n) m = fmaxf(m, slog[n]);
    float s = 0.f;
    for (int n = 0; n < NCOND; ++n) { float e = __expf(slog[n] - m); sw[n] = e; s += e; }
    float inv = 1.f / s;
    for (int n = 0; n < NCOND; ++n) sw[n] *= inv;
  }
  __syncthreads();

  // effective filters (write every element incl. cin pad -> ws fully initialized)
  for (int idx = tid; idx < WPB; idx += 256) {
    const int co  = idx / (9 * KP);
    const int rem = idx % (9 * KP);
    const int tap = rem / KP;
    const int ci  = rem % KP;
    float acc = 0.f;
    if (ci < CIN_) {
      const int kh = tap / 3, kw = tap % 3;
      const int base = ((co * CIN_ + ci) * 3 + kh) * 3 + kw;
      #pragma unroll
      for (int n = 0; n < NCOND; ++n)
        acc += sw[n] * filt_w[n * (COUT_ * CIN_ * 9) + base];
    }
    wEff[(size_t)b * WPB + idx] = (_Float16)acc;
  }
  if (tid < COUT_) {
    float acc = 0.f;
    #pragma unroll
    for (int n = 0; n < NCOND; ++n) acc += sw[n] * filt_b[n * COUT_ + tid];
    bEff[b * COUT_ + tid] = acc;
  }
}

// ---------------------------------------------------------------------------
// Kernel B: per-(batch,row) 3x3 conv as 9 shifted WMMA GEMMs
//   grid = (H, B), 512 threads = 16 wave32; each wave: 1 M-tile x 2 N-tiles
// ---------------------------------------------------------------------------
__global__ __launch_bounds__(TPB) void dfc_conv(
    const float* __restrict__ x,      // [B, CIN, H, W] f32
    const _Float16* __restrict__ wEff,
    const float* __restrict__ bEff,
    float* __restrict__ out) {        // [B, COUT, H, W] f32
  const int h = blockIdx.x;
  const int b = blockIdx.y;
  const int tid = threadIdx.x;

  __shared__ __align__(16) _Float16 sW[WPB];          // 82944 B  [cout][tap][KP]
  __shared__ __align__(16) _Float16 sX[3 * 130 * KP]; // 56160 B  [row][col(130)][KP]

  // weights for this batch: straight b128 copy global -> LDS
  {
    const uint4* src = (const uint4*)(wEff + (size_t)b * WPB);
    uint4* dst = (uint4*)sW;
    const int n16 = WPB / 8; // 5184
    for (int i = tid; i < n16; i += TPB) dst[i] = src[i];
  }
  // zero the halo columns (col 0 and col 129) for all 3 rows
  for (int p = tid; p < 3 * 2 * CIN_; p += TPB) {
    const int row = p / (2 * CIN_);
    const int r2  = p % (2 * CIN_);
    const int col = (r2 < CIN_) ? 0 : 129;
    const int ci  = r2 % CIN_;
    sX[(row * 130 + col) * KP + ci] = (_Float16)0.f;
  }
  // interior fill, transposing to [row][col][cin]; OOB rows -> zeros
  {
    const int chunks = 3 * CIN_ * (W_ / 4); // 6144 float4 chunks
    for (int c = tid; c < chunks; c += TPB) {
      const int row = c / (CIN_ * (W_ / 4));
      const int rem = c % (CIN_ * (W_ / 4));
      const int ci  = rem / (W_ / 4);
      const int w4  = rem % (W_ / 4);
      const int hin = h + row - 1;
      float4 v = make_float4(0.f, 0.f, 0.f, 0.f);
      if (hin >= 0 && hin < H_)
        v = *(const float4*)(x + (((size_t)b * CIN_ + ci) * H_ + hin) * W_ + w4 * 4);
      const int colb = 1 + w4 * 4;
      sX[(row * 130 + colb + 0) * KP + ci] = (_Float16)v.x;
      sX[(row * 130 + colb + 1) * KP + ci] = (_Float16)v.y;
      sX[(row * 130 + colb + 2) * KP + ci] = (_Float16)v.z;
      sX[(row * 130 + colb + 3) * KP + ci] = (_Float16)v.w;
    }
  }
  __syncthreads();

  const int wave  = tid >> 5;
  const int lane  = tid & 31;
  const int hlf   = lane >> 4;       // lane half (0: lanes 0-15, 1: lanes 16-31)
  const int l15   = lane & 15;
  const int mbase = (wave & 3) * 16;          // 4 M-tiles over COUT
  const int nbase = (wave >> 2) * 32;         // each wave: 2 N-tiles of 16 pixels

  // per-lane bases; all tap/tile/k variation folds into DS immediate offsets
  const _Float16* wb = &sW[(mbase + l15) * 9 * KP + hlf * 16];
  const _Float16* xb = &sX[(nbase + l15) * KP + hlf * 16];

  f8v acc[2];
  #pragma unroll
  for (int t = 0; t < 2; ++t)
    #pragma unroll
    for (int r = 0; r < 8; ++r) acc[t][r] = 0.f;

  #pragma unroll
  for (int tap = 0; tap < 9; ++tap) {
    const int kh = tap / 3, kw = tap % 3;
    #pragma unroll
    for (int kb = 0; kb < 2; ++kb) {
      // A fragment (16x32 f16): elems 0-15 = K[k0+16h .. +15] contiguous
      // (matches B)  -> two aligned 16B LDS loads
      const h8 a0 = *(const h8*)(wb + tap * KP + kb * 32);
      const h8 a1 = *(const h8*)(wb + tap * KP + kb * 32 + 8);
      const h16v afrag = __builtin_shufflevector(
          a0, a1, 0, 1, 2, 3, 4, 5, 6, 7, 8, 9, 10, 11, 12, 13, 14, 15);
      #pragma unroll
      for (int t = 0; t < 2; ++t) {
        // B fragment (32x16 f16): N = lane&15, K = 16*half + 0..15 contiguous
        const _Float16* xc = xb + (kh * 130 + kw + t * 16) * KP + kb * 32;
        const h8 b0 = *(const h8*)(xc);
        const h8 b1 = *(const h8*)(xc + 8);
        const h16v bfrag = __builtin_shufflevector(
            b0, b1, 0, 1, 2, 3, 4, 5, 6, 7, 8, 9, 10, 11, 12, 13, 14, 15);
        acc[t] = __builtin_amdgcn_wmma_f32_16x16x32_f16(
            /*neg_a=*/false, afrag, /*neg_b=*/false, bfrag,
            /*c_mod=*/(short)0, acc[t], /*reuse_a=*/false, /*reuse_b=*/false);
      }
    }
  }

  // epilogue: bias + store. D layout: VGPR r -> M = r + 8*half, N = lane&15
  float bias[8];
  #pragma unroll
  for (int r = 0; r < 8; ++r)
    bias[r] = bEff[b * COUT_ + mbase + r + hlf * 8];
  #pragma unroll
  for (int t = 0; t < 2; ++t) {
    const int col = nbase + t * 16 + l15;
    #pragma unroll
    for (int r = 0; r < 8; ++r) {
      const int m = mbase + r + hlf * 8;
      out[(((size_t)b * COUT_ + m) * H_ + h) * W_ + col] = acc[t][r] + bias[r];
    }
  }
}

// ---------------------------------------------------------------------------
extern "C" void kernel_launch(void* const* d_in, const int* in_sizes, int n_in,
                              void* d_out, int out_size, void* d_ws, size_t ws_size,
                              hipStream_t stream) {
  const float* x         = (const float*)d_in[0]; // [16,64,128,128]
  const float* condition = (const float*)d_in[1]; // [16,64]
  const float* filt_w    = (const float*)d_in[2]; // [8,64,64,3,3]
  const float* filt_b    = (const float*)d_in[3]; // [8,64]
  const float* sel_w     = (const float*)d_in[4]; // [8,64]
  const float* sel_b     = (const float*)d_in[5]; // [8]
  float* out = (float*)d_out;

  _Float16* wEff = (_Float16*)d_ws;                               // 16*41472*2 B
  float*    bEff = (float*)((char*)d_ws + (size_t)B_ * WPB * 2);  // +4 KB

  dfc_weights<<<B_, 256, 0, stream>>>(condition, filt_w, filt_b, sel_w, sel_b,
                                      wEff, bEff);
  dim3 grid(H_, B_);
  dfc_conv<<<grid, TPB, 0, stream>>>(x, wEff, bEff, out);
}
